// HistogramWeightedBCELoss_70102456206028
// MI455X (gfx1250) — compile-verified, matrix operand order checked by reading, not run on hardware
//
#include <hip/hip_runtime.h>
#include <stdint.h>

// CDNA5 / gfx1250: wave32 only.
#define WAVE 32

typedef __attribute__((ext_vector_type(8))) int v8i;

// ---------------------------------------------------------------------------
// Workspace layout (unsigned counters):
//   ws[  0.. 63] : histogram of hamming distance (bin = clip(dist,0,63))
//   ws[ 64..127] : n1[j]  = #rows with pred[:,j]==1
//   ws[128..191] : t[j]   = #rows with pred[:,j]==1 && gt[:,j]==1
// ---------------------------------------------------------------------------

__global__ void zero_ws_kernel(unsigned* __restrict__ ws) {
    int t = threadIdx.x;
    if (t < 192) ws[t] = 0u;
}

__device__ __forceinline__ unsigned pack4(unsigned a, unsigned b, unsigned c, unsigned d) {
    return a | (b << 8) | (c << 16) | (d << 24);
}

// Process 8 consecutive columns: produce packed diff bytes (for WMMA A),
// packed pred bytes and packed pred&gt bytes (for column count accumulators).
__device__ __forceinline__ void seg8(float4 p0, float4 p1, float4 g0, float4 g1,
                                     unsigned& dLo, unsigned& dHi,
                                     unsigned& pLo, unsigned& pHi,
                                     unsigned& tLo, unsigned& tHi) {
    // pred/gt are exactly 0.0f or 1.0f -> integer convert is exact.
    unsigned pA = (unsigned)p0.x, pB = (unsigned)p0.y, pC = (unsigned)p0.z, pD = (unsigned)p0.w;
    unsigned pE = (unsigned)p1.x, pF = (unsigned)p1.y, pG = (unsigned)p1.z, pH = (unsigned)p1.w;
    unsigned gA = (unsigned)g0.x, gB = (unsigned)g0.y, gC = (unsigned)g0.z, gD = (unsigned)g0.w;
    unsigned gE = (unsigned)g1.x, gF = (unsigned)g1.y, gG = (unsigned)g1.z, gH = (unsigned)g1.w;
    dLo = pack4(pA ^ gA, pB ^ gB, pC ^ gC, pD ^ gD);
    dHi = pack4(pE ^ gE, pF ^ gF, pG ^ gG, pH ^ gH);
    pLo = pack4(pA, pB, pC, pD);
    pHi = pack4(pE, pF, pG, pH);
    tLo = pack4(pA & gA, pB & gB, pC & gC, pD & gD);
    tHi = pack4(pE & gE, pF & gF, pG & gG, pH & gH);
}

__global__ __launch_bounds__(256) void hamming_hist_colsum_kernel(
    const float* __restrict__ pred, const float* __restrict__ gt,
    unsigned* __restrict__ gws, int nChunks /* rows/16 */, int totalWaves) {
    __shared__ unsigned sh[192];
    int tid = threadIdx.x;
    for (int i = tid; i < 192; i += blockDim.x) sh[i] = 0u;
    __syncthreads();

    unsigned* shist = sh;
    unsigned* sn1   = sh + 64;
    unsigned* st    = sh + 128;

    int lane  = tid & (WAVE - 1);
    int half  = lane >> 4;      // which K-halves of the row this lane owns
    int m     = lane & 15;      // row within the 16-row tile
    int gwave = blockIdx.x * (blockDim.x / WAVE) + (tid / WAVE);

    v8i bOnes, cZero;
#pragma unroll
    for (int i = 0; i < 8; ++i) { bOnes[i] = 0x01010101; cZero[i] = 0; }

    // Packed per-byte column accumulators (each byte counts one column, <=128
    // increments between flushes so no cross-byte carry).
    unsigned accP[8], accT[8];
#pragma unroll
    for (int i = 0; i < 8; ++i) { accP[i] = 0u; accT[i] = 0u; }
    int sinceFlush = 0;

    for (int chunk = gwave; chunk < nChunks; chunk += totalWaves) {
        size_t row = (size_t)chunk * 16 + (size_t)m;
        const float* prow = pred + row * 64;
        const float* grow = gt   + row * 64;

        v8i a;
#pragma unroll
        for (int s = 0; s < 4; ++s) {
            int kbase = s * 16 + half * 8;   // matches 8-bit WMMA A 16x64 layout
            float4 p0 = *(const float4*)(prow + kbase);
            float4 p1 = *(const float4*)(prow + kbase + 4);
            float4 g0 = *(const float4*)(grow + kbase);
            float4 g1 = *(const float4*)(grow + kbase + 4);
            unsigned dLo, dHi, pLo, pHi, tLo, tHi;
            seg8(p0, p1, g0, g1, dLo, dHi, pLo, pHi, tLo, tHi);
            a[2 * s]     = (int)dLo;
            a[2 * s + 1] = (int)dHi;
            accP[2 * s]     += pLo;  accP[2 * s + 1] += pHi;
            accT[2 * s]     += tLo;  accT[2 * s + 1] += tHi;
        }

        // D[m][n] = sum_k diff(row m, k) for every n: 16 hamming distances/wave.
        v8i d = __builtin_amdgcn_wmma_i32_16x16x64_iu8(
            /*sgn_a=*/false, a, /*sgn_b=*/false, bOnes, cZero,
            /*reuse_a=*/false, /*reuse_b=*/false);

        // Pin WMMA result before divergent use so the WMMA cannot be sunk into
        // the EXEC!=all-ones region (WMMA requires full EXEC).
        asm volatile("" :: "v"(d[0]), "v"(d[7]));

        if (m == 0) {  // lanes 0 (rows 0..7) and 16 (rows 8..15)
#pragma unroll
            for (int r = 0; r < 8; ++r) {
                int dist = d[r];
                int bin  = dist > 63 ? 63 : (dist < 0 ? 0 : dist);
                atomicAdd(&shist[bin], 1u);
            }
        }

        if (++sinceFlush >= 128) {  // avoid byte-counter overflow
#pragma unroll
            for (int s = 0; s < 4; ++s) {
                int kbase = s * 16 + half * 8;
#pragma unroll
                for (int j = 0; j < 8; ++j) {
                    unsigned vp = (accP[2 * s + (j >> 2)] >> (8 * (j & 3))) & 0xFFu;
                    unsigned vt = (accT[2 * s + (j >> 2)] >> (8 * (j & 3))) & 0xFFu;
                    atomicAdd(&sn1[kbase + j], vp);
                    atomicAdd(&st [kbase + j], vt);
                }
            }
#pragma unroll
            for (int i = 0; i < 8; ++i) { accP[i] = 0u; accT[i] = 0u; }
            sinceFlush = 0;
        }
    }

    // Final flush of packed column accumulators into LDS.
#pragma unroll
    for (int s = 0; s < 4; ++s) {
        int kbase = s * 16 + half * 8;
#pragma unroll
        for (int j = 0; j < 8; ++j) {
            unsigned vp = (accP[2 * s + (j >> 2)] >> (8 * (j & 3))) & 0xFFu;
            unsigned vt = (accT[2 * s + (j >> 2)] >> (8 * (j & 3))) & 0xFFu;
            atomicAdd(&sn1[kbase + j], vp);
            atomicAdd(&st [kbase + j], vt);
        }
    }
    __syncthreads();
    for (int i = tid; i < 192; i += blockDim.x) atomicAdd(&gws[i], sh[i]);
}

__global__ void finalize_kernel(const unsigned* __restrict__ ws,
                                float* __restrict__ out, int nRows) {
    __shared__ float partial[64];
    int j = threadIdx.x;
    if (j < 64) {
        float h  = (float)ws[j];
        float hs = fminf(h, 0.51f - h);
        float w  = expf(3.0f * hs);           // 1 for empty bins, ~0 for occupied
        float n1 = (float)ws[64 + j];
        float tt = (float)ws[128 + j];
        const float LN2 = 0.69314718055994530942f;
        float c1 = 1.0f + log1pf(expf(-1.0f)); // loss when pred=1, gt=0
        float S  = LN2 * ((float)nRows - n1) + c1 * n1 - tt;
        partial[j] = w * S;
    }
    __syncthreads();
    if (j == 0) {
        float s = 0.0f;
        for (int i = 0; i < 64; ++i) s += partial[i];
        out[0] = s / ((float)nRows * 64.0f);
    }
}

extern "C" void kernel_launch(void* const* d_in, const int* in_sizes, int n_in,
                              void* d_out, int out_size, void* d_ws, size_t ws_size,
                              hipStream_t stream) {
    const float* pred = (const float*)d_in[0];
    const float* gt   = (const float*)d_in[1];
    float*    out = (float*)d_out;
    unsigned* ws  = (unsigned*)d_ws;

    long total  = (long)in_sizes[0];
    int  nRows  = (int)(total / 64);
    int  nChunks = nRows / 16;          // 16 rows per wave per WMMA

    const int threads = 256;            // 8 wave32 per block
    const int blocks  = 512;            // grid-stride; 16 iters/wave at N=1M
    const int totalWaves = blocks * (threads / WAVE);

    zero_ws_kernel<<<1, 256, 0, stream>>>(ws);
    hamming_hist_colsum_kernel<<<blocks, threads, 0, stream>>>(
        pred, gt, ws, nChunks, totalWaves);
    finalize_kernel<<<1, 64, 0, stream>>>(ws, out, nRows);
}
